// SemSegFPNHead_DFConv_87703232184534
// MI455X (gfx1250) — matrix operand order repaired and checked by reading, not compiled, not run on hardware
//
#include <hip/hip_runtime.h>
#include <hip/hip_bf16.h>
#include <math.h>

// ---------------------------------------------------------------------------
// CDNA5 (gfx1250) wave32 WMMA types
// ---------------------------------------------------------------------------
typedef __attribute__((ext_vector_type(16))) _Float16 v16h;
typedef __attribute__((ext_vector_type(8)))  _Float16 v8h;
typedef __attribute__((ext_vector_type(8)))  float    v8f;

struct __align__(16) Smp {
    int   idx[4];   // clamped corner pixel indices (y*W+x)
    float w[4];     // bilinear weight * sigmoid(mask) * validity
};

// ---------------------------------------------------------------------------
// NCHW (C,H,W) -> pixel-major (H*W, C)
// ---------------------------------------------------------------------------
__global__ void chw_to_pc(const float* __restrict__ x, float* __restrict__ xpc,
                          int C, int HW) {
    int tid = blockIdx.x * blockDim.x + threadIdx.x;
    if (tid >= C * HW) return;
    int c = tid / HW;
    int p = tid - c * HW;
    xpc[(size_t)p * C + c] = x[(size_t)c * HW + p];
}

// ---------------------------------------------------------------------------
// Identity sampling table: turns the plain 3x3 offset conv (zero padding)
// into a deformable conv with weight 1 on the in-bounds tap. All four corner
// indices alias the same pixel so the 4-corner gather costs one cache line.
// ---------------------------------------------------------------------------
__global__ void sample_identity(Smp* __restrict__ smp, int H, int W) {
    int tid = blockIdx.x * blockDim.x + threadIdx.x;
    int k = tid % 9;
    int p = tid / 9;
    if (p >= H * W) return;
    int y = p / W, x = p - y * W;
    int py = y + (k / 3) - 1;
    int px = x + (k % 3) - 1;
    bool valid = (py >= 0) && (py < H) && (px >= 0) && (px < W);
    int idx = min(max(py, 0), H - 1) * W + min(max(px, 0), W - 1);
    Smp s;
#pragma unroll
    for (int i = 0; i < 4; ++i) { s.idx[i] = idx; s.w[i] = 0.0f; }
    s.w[0] = valid ? 1.0f : 0.0f;
    smp[(size_t)p * 9 + k] = s;
}

// ---------------------------------------------------------------------------
// Per-(pixel, tap) sampling prep from offset-conv output om[Npix, 32]:
// fold sigmoid(mask) * bilinear weights * validity.
// ---------------------------------------------------------------------------
__global__ void sample_prep(const float* __restrict__ om, Smp* __restrict__ smp,
                            int H, int W) {
    int tid = blockIdx.x * blockDim.x + threadIdx.x;
    int k = tid % 9;
    int p = tid / 9;
    if (p >= H * W) return;
    int y = p / W, x = p - y * W;
    const float* o = om + (size_t)p * 32;
    float dy = o[2 * k], dx = o[2 * k + 1];
    float m  = 1.0f / (1.0f + expf(-o[18 + k]));
    float py = (float)y + (float)(k / 3 - 1) + dy;
    float px = (float)x + (float)(k % 3 - 1) + dx;
    float y0f = floorf(py), x0f = floorf(px);
    float wy1 = py - y0f, wx1 = px - x0f;
    int y0 = (int)y0f, x0 = (int)x0f;
    Smp s;
#pragma unroll
    for (int i = 0; i < 4; ++i) {
        int yc = y0 + (i >> 1);
        int xc = x0 + (i & 1);
        float wgt = ((i >> 1) ? wy1 : (1.0f - wy1)) * ((i & 1) ? wx1 : (1.0f - wx1));
        bool valid = (yc >= 0) && (yc < H) && (xc >= 0) && (xc < W);
        int yi = min(max(yc, 0), H - 1);
        int xi = min(max(xc, 0), W - 1);
        s.idx[i] = yi * W + xi;
        s.w[i]   = valid ? wgt * m : 0.0f;
    }
    smp[(size_t)p * 9 + k] = s;
}

// ---------------------------------------------------------------------------
// Repack conv weights [O_real,C,3,3] f32 -> f16 [O_pad, K=9C], kk = k*C+c,
// zero rows for o >= O_real.
// ---------------------------------------------------------------------------
__global__ void repack_w_pad(const float* __restrict__ w, _Float16* __restrict__ wt,
                             int O_real, int O_pad, int C) {
    int K = 9 * C;
    int tid = blockIdx.x * blockDim.x + threadIdx.x;
    if (tid >= O_pad * K) return;
    int o = tid / K;
    int kk = tid - o * K;
    int k = kk / C;
    int c = kk - k * C;
    wt[tid] = (_Float16)((o < O_real)
                             ? w[(size_t)o * C * 9 + (size_t)c * 9 + k]
                             : 0.0f);
}

__global__ void repack_pred(const float* __restrict__ w, _Float16* __restrict__ wt) {
    int tid = blockIdx.x * blockDim.x + threadIdx.x;   // 64*512
    if (tid >= 64 * 512) return;
    int o = tid / 512;
    int c = tid - o * 512;
    wt[tid] = (_Float16)((o < 54) ? w[o * 512 + c] : 0.0f);
}

// ---------------------------------------------------------------------------
// A-fragment load from LDS tile row (16-bit A layout, ISA 7.12.2):
// lane L: M = L%16, halves K = {b..b+7, b+16..b+23}, b = 8*(L>=16).
// ---------------------------------------------------------------------------
__device__ __forceinline__ v16h ld_frag_a(const _Float16* row, int kb) {
    v8h lo = *(const v8h*)(row + kb);
    v8h hi = *(const v8h*)(row + kb + 16);
    v16h a;
#pragma unroll
    for (int i = 0; i < 8; ++i) { a[i] = lo[i]; a[8 + i] = hi[i]; }
    return a;
}

// ---------------------------------------------------------------------------
// Fused deformable-im2col + WMMA GEMM (also runs the offset convs via
// identity sampling). Block: 256 threads = 8 waves, M-tile = 128 pixels,
// K stepped by 32 (one tap x 32 channels per step since C % 32 == 0).
// Gather is float4-vectorized (global_load_b128 over contiguous channels).
// ---------------------------------------------------------------------------
template <int CIN, int OC, int OC_REAL, bool RELU>
__global__ __launch_bounds__(256) void dcn_gemm(
    const float* __restrict__ xpc, const Smp* __restrict__ smp,
    const _Float16* __restrict__ wt, const float* __restrict__ bias,
    float* __restrict__ out, int out_pitch, int out_col0) {
    constexpr int K = 9 * CIN;
    __shared__ __align__(64) _Float16 Atile[128][32];
    __shared__ __align__(64) _Float16 Btile[OC][32];

    const int tid  = threadIdx.x;
    const int lane = tid & 31;
    const int wave = tid >> 5;            // 0..7 -> M sub-tile
    const int pblk = blockIdx.x * 128;
    const int ap   = tid >> 1;            // pixel 0..127 for A fill
    const int ac0  = (tid & 1) * 16;      // channel half for A fill

    v8f acc[OC / 16];
#pragma unroll
    for (int n = 0; n < OC / 16; ++n) acc[n] = (v8f)(0.0f);

    const int colL = lane & 15;
    const int kbA  = (lane < 16) ? 0 : 8;    // A K-base (interleaved halves)
    const int kbB  = (lane < 16) ? 0 : 16;   // unused: B frag is loaded with kbA (interleaved halves) too

    for (int k = 0; k < 9; ++k) {
        Smp s = smp[(size_t)(pblk + ap) * 9 + k];
        const float4* x0v = (const float4*)(xpc + (size_t)s.idx[0] * CIN);
        const float4* x1v = (const float4*)(xpc + (size_t)s.idx[1] * CIN);
        const float4* x2v = (const float4*)(xpc + (size_t)s.idx[2] * CIN);
        const float4* x3v = (const float4*)(xpc + (size_t)s.idx[3] * CIN);

        for (int cc = 0; cc < CIN; cc += 32) {
            // ---- cooperative A fill: vectorized deformable gather -> f16 ----
            alignas(16) _Float16 av[16];
            const int c4 = (cc + ac0) >> 2;
#pragma unroll
            for (int j4 = 0; j4 < 4; ++j4) {
                float4 a0 = x0v[c4 + j4];
                float4 a1 = x1v[c4 + j4];
                float4 a2 = x2v[c4 + j4];
                float4 a3 = x3v[c4 + j4];
                av[4 * j4 + 0] = (_Float16)(s.w[0] * a0.x + s.w[1] * a1.x +
                                            s.w[2] * a2.x + s.w[3] * a3.x);
                av[4 * j4 + 1] = (_Float16)(s.w[0] * a0.y + s.w[1] * a1.y +
                                            s.w[2] * a2.y + s.w[3] * a3.y);
                av[4 * j4 + 2] = (_Float16)(s.w[0] * a0.z + s.w[1] * a1.z +
                                            s.w[2] * a2.z + s.w[3] * a3.z);
                av[4 * j4 + 3] = (_Float16)(s.w[0] * a0.w + s.w[1] * a1.w +
                                            s.w[2] * a2.w + s.w[3] * a3.w);
            }
            *(v8h*)&Atile[ap][ac0]     = *(const v8h*)&av[0];
            *(v8h*)&Atile[ap][ac0 + 8] = *(const v8h*)&av[8];

            // ---- cooperative B fill: 32 contiguous K halves per out-chan ----
            if (tid < OC) {
                const v8h* src = (const v8h*)(wt + (size_t)tid * K + k * CIN + cc);
                v8h* dst = (v8h*)&Btile[tid][0];
                dst[0] = src[0]; dst[1] = src[1]; dst[2] = src[2]; dst[3] = src[3];
                __builtin_prefetch(src + 8, 0, 0);   // global_prefetch next chunk
            }
            __syncthreads();

            // ---- WMMA: one A frag per wave, loop over N tiles ----
            v16h a = ld_frag_a(&Atile[wave * 16 + colL][0], kbA);
#pragma unroll
            for (int n = 0; n < OC / 16; ++n) {
                v16h b = ld_frag_a(&Btile[n * 16 + colL][0], kbA);
                acc[n] = __builtin_amdgcn_wmma_f32_16x16x32_f16(
                    false, a, false, b, (short)0, acc[n], false, false);
            }
            __syncthreads();
        }
    }

    // ---- epilogue: bias / ReLU, coalesced pixel-major store ----
    const int rowoff = (lane < 16) ? 0 : 8;
#pragma unroll
    for (int n = 0; n < OC / 16; ++n) {
        int o = n * 16 + colL;
        bool active = (OC_REAL >= OC) || (o < OC_REAL);
        float badd = (bias != nullptr && active) ? bias[o] : 0.0f;
#pragma unroll
        for (int r = 0; r < 8; ++r) {
            float v = acc[n][r] + badd;
            if (RELU) v = v > 0.0f ? v : 0.0f;
            if (active) {
                int p = pblk + wave * 16 + r + rowoff;
                out[(size_t)p * out_pitch + out_col0 + o] = v;
            }
        }
    }
}

// ---------------------------------------------------------------------------
// 1x1 pred conv as WMMA GEMM: feat[16384,512] x w[64(pad),512] + bias -> [16384,64]
// ---------------------------------------------------------------------------
__global__ __launch_bounds__(256) void pred_gemm(
    const float* __restrict__ feat, const _Float16* __restrict__ wt,
    const float* __restrict__ bias, float* __restrict__ outl) {
    constexpr int K = 512, OC = 64;
    __shared__ __align__(64) _Float16 Atile[128][32];
    __shared__ __align__(64) _Float16 Btile[OC][32];

    const int tid  = threadIdx.x;
    const int lane = tid & 31;
    const int wave = tid >> 5;
    const int pblk = blockIdx.x * 128;
    const int ap   = tid >> 1;
    const int ac0  = (tid & 1) * 16;
    const int colL = lane & 15;
    const int kbA  = (lane < 16) ? 0 : 8;
    const int kbB  = (lane < 16) ? 0 : 16;

    v8f acc[OC / 16];
#pragma unroll
    for (int n = 0; n < OC / 16; ++n) acc[n] = (v8f)(0.0f);

    for (int kk0 = 0; kk0 < K; kk0 += 32) {
#pragma unroll
        for (int j = 0; j < 16; ++j)
            Atile[ap][ac0 + j] =
                (_Float16)feat[(size_t)(pblk + ap) * K + kk0 + ac0 + j];
        if (tid < OC) {
            const v8h* src = (const v8h*)(wt + (size_t)tid * K + kk0);
            v8h* dst = (v8h*)&Btile[tid][0];
            dst[0] = src[0]; dst[1] = src[1]; dst[2] = src[2]; dst[3] = src[3];
        }
        __syncthreads();
        v16h a = ld_frag_a(&Atile[wave * 16 + colL][0], kbA);
#pragma unroll
        for (int n = 0; n < OC / 16; ++n) {
            v16h b = ld_frag_a(&Btile[n * 16 + colL][0], kbA);
            acc[n] = __builtin_amdgcn_wmma_f32_16x16x32_f16(
                false, a, false, b, (short)0, acc[n], false, false);
        }
        __syncthreads();
    }

    const int rowoff = (lane < 16) ? 0 : 8;
#pragma unroll
    for (int n = 0; n < OC / 16; ++n) {
        int o = n * 16 + colL;
        float b = (o < 54) ? bias[o] : 0.0f;
#pragma unroll
        for (int r = 0; r < 8; ++r) {
            if (o < 54) {
                int p = pblk + wave * 16 + r + rowoff;
                outl[(size_t)p * 64 + o] = acc[n][r] + b;
            }
        }
    }
}

// ---------------------------------------------------------------------------
// Bilinear upsample (align_corners=True) level [H*W,128] -> feat[16384,512] cols
// ---------------------------------------------------------------------------
__global__ void upsample_ac_concat(const float* __restrict__ lvl,
                                   float* __restrict__ feat,
                                   int H, int W, int coloff) {
    int tid = blockIdx.x * blockDim.x + threadIdx.x;
    if (tid >= 16384 * 128) return;
    int c = tid & 127;
    int p = tid >> 7;
    int yo = p >> 7, xo = p & 127;
    float ys = (float)yo * ((float)(H - 1) / 127.0f);
    float xs = (float)xo * ((float)(W - 1) / 127.0f);
    int y0 = (int)floorf(ys), x0 = (int)floorf(xs);
    int y1 = min(y0 + 1, H - 1), x1 = min(x0 + 1, W - 1);
    float wy = ys - (float)y0, wx = xs - (float)x0;
    float v00 = lvl[(size_t)(y0 * W + x0) * 128 + c];
    float v01 = lvl[(size_t)(y0 * W + x1) * 128 + c];
    float v10 = lvl[(size_t)(y1 * W + x0) * 128 + c];
    float v11 = lvl[(size_t)(y1 * W + x1) * 128 + c];
    float v = v00 * (1 - wy) * (1 - wx) + v01 * (1 - wy) * wx
            + v10 * wy * (1 - wx) + v11 * wy * wx;
    feat[(size_t)p * 512 + coloff + c] = v;
}

// ---------------------------------------------------------------------------
// Final x4 upsample (align_corners=False): logits[16384,64] -> out[54,512,512]
// ---------------------------------------------------------------------------
__global__ void upsample_final(const float* __restrict__ logits,
                               float* __restrict__ out) {
    int tid = blockIdx.x * blockDim.x + threadIdx.x;
    if (tid >= 54 * 512 * 512) return;
    int xo = tid & 511;
    int yo = (tid >> 9) & 511;
    int co = tid >> 18;
    float ys = fmaxf(((float)yo + 0.5f) * 0.25f - 0.5f, 0.0f);
    float xs = fmaxf(((float)xo + 0.5f) * 0.25f - 0.5f, 0.0f);
    int y0 = (int)ys, x0 = (int)xs;     // floor of non-negative
    int y1 = min(y0 + 1, 127), x1 = min(x0 + 1, 127);
    float wy = ys - (float)y0, wx = xs - (float)x0;
    float v00 = logits[(size_t)(y0 * 128 + x0) * 64 + co];
    float v01 = logits[(size_t)(y0 * 128 + x1) * 64 + co];
    float v10 = logits[(size_t)(y1 * 128 + x0) * 64 + co];
    float v11 = logits[(size_t)(y1 * 128 + x1) * 64 + co];
    out[tid] = v00 * (1 - wy) * (1 - wx) + v01 * (1 - wy) * wx
             + v10 * wy * (1 - wx) + v11 * wy * wx;
}

// ---------------------------------------------------------------------------
// Host orchestration
// ---------------------------------------------------------------------------
static inline int cdiv(int a, int b) { return (a + b - 1) / b; }

extern "C" void kernel_launch(void* const* d_in, const int* in_sizes, int n_in,
                              void* d_out, int out_size, void* d_ws, size_t ws_size,
                              hipStream_t stream) {
    (void)in_sizes; (void)n_in; (void)out_size; (void)ws_size;
    const float* pin[4]   = {(const float*)d_in[0], (const float*)d_in[1],
                             (const float*)d_in[2], (const float*)d_in[3]};
    const float* w_off[3] = {(const float*)d_in[4], (const float*)d_in[7],
                             (const float*)d_in[10]};
    const float* b_off[3] = {(const float*)d_in[5], (const float*)d_in[8],
                             (const float*)d_in[11]};
    const float* w_conv[3]= {(const float*)d_in[6], (const float*)d_in[9],
                             (const float*)d_in[12]};
    const float* w_pred   = (const float*)d_in[13];
    const float* b_pred   = (const float*)d_in[14];
    float* out = (float*)d_out;

    // Workspace bump allocator (deterministic layout, ~85 MB)
    char* ws = (char*)d_ws;
    auto alloc = [&](size_t bytes) -> void* {
        void* p = ws;
        ws += (bytes + 255) & ~(size_t)255;
        return p;
    };
    _Float16* wt0   = (_Float16*)alloc((size_t)256 * 2304 * 2);
    _Float16* wt1   = (_Float16*)alloc((size_t)128 * 2304 * 2);
    _Float16* wt2   = (_Float16*)alloc((size_t)128 * 1152 * 2);
    _Float16* wto0  = (_Float16*)alloc((size_t)32 * 2304 * 2);
    _Float16* wto1  = (_Float16*)alloc((size_t)32 * 2304 * 2);
    _Float16* wto2  = (_Float16*)alloc((size_t)32 * 1152 * 2);
    _Float16* wtp   = (_Float16*)alloc((size_t)64 * 512 * 2);
    float* bufX     = (float*)alloc((size_t)16384 * 256 * 4);
    float* bufY     = (float*)alloc((size_t)16384 * 256 * 4);
    float* om       = (float*)alloc((size_t)16384 * 32 * 4);
    Smp*   smp      = (Smp*)  alloc((size_t)16384 * 9 * sizeof(Smp));
    Smp*   smpI     = (Smp*)  alloc((size_t)16384 * 9 * sizeof(Smp));
    float* feat     = (float*)alloc((size_t)16384 * 512 * 4);
    float* logits   = (float*)alloc((size_t)16384 * 64 * 4);

    // One-time weight repacks (f32 OIHW -> f16 [O_pad, 9C], kk = k*C+c)
    repack_w_pad<<<cdiv(256 * 2304, 256), 256, 0, stream>>>(w_conv[0], wt0, 256, 256, 256);
    repack_w_pad<<<cdiv(128 * 2304, 256), 256, 0, stream>>>(w_conv[1], wt1, 128, 128, 256);
    repack_w_pad<<<cdiv(128 * 1152, 256), 256, 0, stream>>>(w_conv[2], wt2, 128, 128, 128);
    repack_w_pad<<<cdiv(32 * 2304, 256), 256, 0, stream>>>(w_off[0], wto0, 27, 32, 256);
    repack_w_pad<<<cdiv(32 * 2304, 256), 256, 0, stream>>>(w_off[1], wto1, 27, 32, 256);
    repack_w_pad<<<cdiv(32 * 1152, 256), 256, 0, stream>>>(w_off[2], wto2, 27, 32, 128);
    repack_pred<<<cdiv(64 * 512, 256), 256, 0, stream>>>(w_pred, wtp);

    const int Hs[4] = {128, 64, 32, 16};
    for (int l = 0; l < 4; ++l) {
        const int H = Hs[l], W = Hs[l], NP = H * W;
        chw_to_pc<<<cdiv(256 * NP, 256), 256, 0, stream>>>(pin[l], bufX, 256, NP);
        sample_identity<<<cdiv(NP * 9, 256), 256, 0, stream>>>(smpI, H, W);

        // layer0: 256 -> 256
        dcn_gemm<256, 32, 27, false><<<NP / 128, 256, 0, stream>>>(
            bufX, smpI, wto0, b_off[0], om, 32, 0);
        sample_prep<<<cdiv(NP * 9, 256), 256, 0, stream>>>(om, smp, H, W);
        dcn_gemm<256, 256, 256, true><<<NP / 128, 256, 0, stream>>>(
            bufX, smp, wt0, nullptr, bufY, 256, 0);

        // layer1: 256 -> 128
        dcn_gemm<256, 32, 27, false><<<NP / 128, 256, 0, stream>>>(
            bufY, smpI, wto1, b_off[1], om, 32, 0);
        sample_prep<<<cdiv(NP * 9, 256), 256, 0, stream>>>(om, smp, H, W);
        dcn_gemm<256, 128, 128, true><<<NP / 128, 256, 0, stream>>>(
            bufY, smp, wt1, nullptr, bufX, 128, 0);

        // layer2: 128 -> 128
        dcn_gemm<128, 32, 27, false><<<NP / 128, 256, 0, stream>>>(
            bufX, smpI, wto2, b_off[2], om, 32, 0);
        sample_prep<<<cdiv(NP * 9, 256), 256, 0, stream>>>(om, smp, H, W);
        if (l == 0) {
            dcn_gemm<128, 128, 128, true><<<NP / 128, 256, 0, stream>>>(
                bufX, smp, wt2, nullptr, feat, 512, 0);
        } else {
            dcn_gemm<128, 128, 128, true><<<NP / 128, 256, 0, stream>>>(
                bufX, smp, wt2, nullptr, bufY, 128, 0);
            upsample_ac_concat<<<cdiv(16384 * 128, 256), 256, 0, stream>>>(
                bufY, feat, H, W, 128 * l);
        }
    }

    pred_gemm<<<16384 / 128, 256, 0, stream>>>(feat, wtp, b_pred, logits);
    upsample_final<<<cdiv(54 * 512 * 512, 256), 256, 0, stream>>>(logits, out);
}
